// ICOSA_backbone_2671469658231
// MI455X (gfx1250) — matrix operand order repaired and checked by reading, not compiled, not run on hardware
//
#include <hip/hip_runtime.h>
#include <hip/hip_bf16.h>

typedef __attribute__((ext_vector_type(16))) _Float16 v16h;
typedef __attribute__((ext_vector_type(8)))  float    v8f;

union HalfPair { unsigned int u; _Float16 h[2]; };

// ---------------------------------------------------------------------------
// WMMA fragment loaders (CDNA5 ISA 7.12.2 layouts, wave32)
//  A 16x32 f16:  lane L (L&15 = row M). VGPR i<4 holds K = 2i,2i+1 (+8 if L>=16)
//                VGPR i>=4 holds K = 16+2(i-4),.. (+8 if L>=16)
//  B 32x16 f16:  lane L (L&15 = col N). VGPR i<4 holds K = 2i,2i+1, i>=4 K = 16+2(i-4),.. (+8 if L>=16)
//  C 16x16 f32:  lane L col = L&15, VGPR r row = r + (L<16 ? 0 : 8)
// ---------------------------------------------------------------------------

// A fragment from f32 global, unconditional clamped loads (rows always valid
// here since R % 16 == 0; ragged K handled by address clamp + zero-padded B).
__device__ __forceinline__ v16h frag_a_f32(const float* __restrict__ X, int Kd,
                                           int rowBase, int kb, int lane) {
    int row  = rowBase + (lane & 15);
    int koff = (lane < 16) ? 0 : 8;
    int kmax = Kd - 2;                      // Kd is always even here
    const float* xr = X + (size_t)row * Kd;
    v16h a;
#pragma unroll
    for (int i = 0; i < 8; ++i) {
        int k = kb + koff + ((i < 4) ? 2 * i : 16 + 2 * (i - 4));
        if (k > kmax) k = kmax;             // clamp: garbage * zero-padded B = 0
        float2 f = *(const float2*)(xr + k);
        a[2 * i]     = (_Float16)f.x;
        a[2 * i + 1] = (_Float16)f.y;
    }
    return a;
}

// A fragment from f16 LDS tile (row-major, stride kpad halves, kpad % 32 == 0).
__device__ __forceinline__ v16h frag_a_ldsh(const _Float16* S, int kpad, int kb, int lane) {
    int row  = lane & 15;
    int koff = (lane < 16) ? 0 : 8;
    const _Float16* sr = S + row * kpad;
    v16h a;
#pragma unroll
    for (int i = 0; i < 8; ++i) {
        int k = kb + koff + ((i < 4) ? 2 * i : 16 + 2 * (i - 4));
        HalfPair cv; cv.u = *(const unsigned int*)(sr + k);
        a[2 * i]     = cv.h[0];
        a[2 * i + 1] = cv.h[1];
    }
    return a;
}

// B fragment from pre-packed f16 weights: WP[p * N + n] holds (k=2p, k=2p+1),
// zero-padded to Kp. 8 unconditional dword loads, no converts.
__device__ __forceinline__ v16h frag_b_packed(const unsigned int* __restrict__ WP, int N,
                                              int kb, int colBase, int lane) {
    int col = colBase + (lane & 15);
    int p0  = (kb + ((lane < 16) ? 0 : 8)) >> 1;
    const unsigned int* wp = WP + (size_t)p0 * N + col;
    v16h b;
#pragma unroll
    for (int i = 0; i < 8; ++i) {
        HalfPair cv; cv.u = wp[(size_t)((i < 4) ? i : i + 4) * N];
        b[2 * i]     = cv.h[0];
        b[2 * i + 1] = cv.h[1];
    }
    return b;
}

__device__ __forceinline__ v8f wmma32(v16h a, v16h b, v8f c) {
    return __builtin_amdgcn_wmma_f32_16x16x32_f16(false, a, false, b, (short)0, c, false, false);
}

// ---------------------------------------------------------------------------
// Weight packer: f32 (Kd,N) -> packed f16 pairs, K zero-padded to Kp.
// ---------------------------------------------------------------------------
__global__ void k_pack_w(const float* __restrict__ W, unsigned int* __restrict__ WP,
                         int Kd, int N, int Kp) {
    int i = blockIdx.x * blockDim.x + threadIdx.x;
    int total = (Kp / 2) * N;
    if (i >= total) return;
    int p = i / N, n = i % N;
    int k0 = 2 * p, k1 = k0 + 1;
    HalfPair cv;
    cv.h[0] = (_Float16)((k0 < Kd) ? W[(size_t)k0 * N + n] : 0.0f);
    cv.h[1] = (_Float16)((k1 < Kd) ? W[(size_t)k1 * N + n] : 0.0f);
    WP[i] = cv.u;
}

// ---------------------------------------------------------------------------
// Generic GEMM, 16x64 tile per wave (A frag reused by 4 WMMAs / k-step).
// ---------------------------------------------------------------------------
template <int ACT, int HAS_BIAS>
__global__ void k_gemm(const float* __restrict__ X, const unsigned int* __restrict__ WP,
                       const float* __restrict__ bias, float* __restrict__ Y,
                       int R, int Kd, int Kp, int N) {
    int lane    = threadIdx.x;
    int rowBase = blockIdx.x * 16;
    int colBase = blockIdx.y * 64;
    __builtin_prefetch(WP, 0, 1);
    v8f c0 = {}, c1 = {}, c2 = {}, c3 = {};
    for (int kb = 0; kb < Kp; kb += 32) {
        v16h a  = frag_a_f32(X, Kd, rowBase, kb, lane);
        v16h b0 = frag_b_packed(WP, N, kb, colBase,      lane);
        v16h b1 = frag_b_packed(WP, N, kb, colBase + 16, lane);
        v16h b2 = frag_b_packed(WP, N, kb, colBase + 32, lane);
        v16h b3 = frag_b_packed(WP, N, kb, colBase + 48, lane);
        c0 = wmma32(a, b0, c0);
        c1 = wmma32(a, b1, c1);
        c2 = wmma32(a, b2, c2);
        c3 = wmma32(a, b3, c3);
    }
    int colL = lane & 15;
    int r0   = rowBase + ((lane < 16) ? 0 : 8);
    float bv0 = 0.f, bv1 = 0.f, bv2 = 0.f, bv3 = 0.f;
    if (HAS_BIAS) {
        bv0 = bias[colBase + colL];
        bv1 = bias[colBase + 16 + colL];
        bv2 = bias[colBase + 32 + colL];
        bv3 = bias[colBase + 48 + colL];
    }
#pragma unroll
    for (int r = 0; r < 8; ++r) {
        int row = r0 + r;
        float v0 = c0[r] + bv0, v1 = c1[r] + bv1, v2 = c2[r] + bv2, v3 = c3[r] + bv3;
        if (ACT) {
            v0 = fmaxf(v0, 0.0f); v1 = fmaxf(v1, 0.0f);
            v2 = fmaxf(v2, 0.0f); v3 = fmaxf(v3, 0.0f);
        }
        float* yr = Y + (size_t)row * N + colBase + colL;
        yr[0]  = v0;
        yr[16] = v1;
        yr[32] = v2;
        yr[48] = v3;
    }
}

// ---------------------------------------------------------------------------
// In-wave GEMM on a 16-row f16 LDS tile against packed weights.
// Two 16-col tiles per A-fragment load (Nout % 32 == 0).
// All epilogue options are compile-time -> straight-line epilogue code.
// ---------------------------------------------------------------------------
template <int ACT, bool HAS_BIAS, bool HAS_AFF, bool OUT_H, bool OUT_F>
__device__ __forceinline__ void lds_gemm_h(const _Float16* S, int kpad,
                                           const unsigned int* __restrict__ WP,
                                           const float* __restrict__ bias, int Nout,
                                           _Float16* Yh, float* Yf,
                                           const float* __restrict__ gamma,
                                           const float* __restrict__ beta,
                                           float inv, int lane) {
    for (int nb = 0; nb < Nout; nb += 32) {
        v8f c0 = {}, c1 = {};
        for (int kb = 0; kb < kpad; kb += 32) {
            v16h a  = frag_a_ldsh(S, kpad, kb, lane);
            v16h b0 = frag_b_packed(WP, Nout, kb, nb,      lane);
            v16h b1 = frag_b_packed(WP, Nout, kb, nb + 16, lane);
            c0 = wmma32(a, b0, c0);
            c1 = wmma32(a, b1, c1);
        }
        int col0 = nb + (lane & 15);
        int col1 = col0 + 16;
        int r0   = (lane < 16) ? 0 : 8;
        float bv0 = 0.f, bv1 = 0.f, gm0 = 0.f, gm1 = 0.f, bt0 = 0.f, bt1 = 0.f;
        if (HAS_BIAS) { bv0 = bias[col0]; bv1 = bias[col1]; }
        if (HAS_AFF)  { gm0 = gamma[col0]; gm1 = gamma[col1];
                        bt0 = beta[col0];  bt1 = beta[col1]; }
#pragma unroll
        for (int r = 0; r < 8; ++r) {
            float v0 = c0[r] + bv0, v1 = c1[r] + bv1;
            if (HAS_AFF) {
                v0 = gm0 * v0 * inv + bt0;
                v1 = gm1 * v1 * inv + bt1;
            }
            if (ACT) { v0 = fmaxf(v0, 0.0f); v1 = fmaxf(v1, 0.0f); }
            if (OUT_H) {
                Yh[(r0 + r) * Nout + col0] = (_Float16)v0;
                Yh[(r0 + r) * Nout + col1] = (_Float16)v1;
            }
            if (OUT_F) {
                Yf[(r0 + r) * Nout + col0] = v0;
                Yf[(r0 + r) * Nout + col1] = v1;
            }
        }
    }
}

// ---------------------------------------------------------------------------
// Fused point-transformer: 1 wave per query point.
// ---------------------------------------------------------------------------
__global__ void k_transformer(const float* __restrict__ xyz, const int* __restrict__ idx,
                              const float* __restrict__ q, const float* __restrict__ kall,
                              const float* __restrict__ vall, const float* __restrict__ pre,
                              const unsigned int* wd0p, const float* bd0,
                              const unsigned int* wd1p, const float* bd1,
                              const unsigned int* wg0p, const float* bg0,
                              const unsigned int* wg1p, const float* bg1,
                              const float* w2, const float* b2,
                              float* __restrict__ out, int Npts, int d_points) {
    __shared__ __align__(4) _Float16 s_deltah[16 * 32];
    __shared__ __align__(4) _Float16 s_t0h[16 * 64];
    __shared__ __align__(4) _Float16 s_tmph[16 * 64];
    __shared__ __align__(4) _Float16 s_gh[16 * 64];
    __shared__ float s_pos[16 * 64];
    __shared__ float s_attn[16 * 64];
    __shared__ float s_vp[16 * 64];
    __shared__ float s_res[64];
    __shared__ int   s_idx[16];

    int lane = threadIdx.x;
    int bm   = blockIdx.x;
    int b    = bm / Npts;
    const float* qm = q + (size_t)bm * 64;

    if (lane < 16) s_idx[lane] = idx[(size_t)bm * 16 + lane];
    __syncthreads();

    float cx = xyz[(size_t)bm * 3 + 0];
    float cy = xyz[(size_t)bm * 3 + 1];
    float cz = xyz[(size_t)bm * 3 + 2];
    for (int t = lane; t < 16 * 32; t += 32) {
        int r = t >> 5, c = t & 31;
        float v = 0.0f;
        if (c < 3) {
            int nb = s_idx[r];
            float ctr = (c == 0) ? cx : (c == 1) ? cy : cz;
            v = ctr - xyz[((size_t)b * Npts + nb) * 3 + c];
        }
        s_deltah[t] = (_Float16)v;
    }
    __syncthreads();

    // pos = fc_delta[1]( relu(fc_delta[0](delta)) )
    lds_gemm_h<1, true, false, true, false>(s_deltah, 32, wd0p, bd0, 64,
                                            s_t0h, nullptr, nullptr, nullptr, 0.0f, lane);
    __syncthreads();
    lds_gemm_h<0, true, false, false, true>(s_t0h, 64, wd1p, bd1, 64,
                                            nullptr, s_pos, nullptr, nullptr, 0.0f, lane);
    __syncthreads();

    // tmp = q - kf + pos (f16 staging) ; vp = v + pos (f32)
    for (int t = lane; t < 16 * 64; t += 32) {
        int r = t >> 6, c = t & 63;
        int nb = s_idx[r];
        size_t gi = ((size_t)b * Npts + nb) * 64 + c;
        s_tmph[t] = (_Float16)(qm[c] - kall[gi] + s_pos[t]);
        s_vp[t]   = vall[gi] + s_pos[t];
    }
    __syncthreads();

    // attn = fc_gamma[1]( relu(fc_gamma[0](tmp)) )
    lds_gemm_h<1, true, false, true, false>(s_tmph, 64, wg0p, bg0, 64,
                                            s_gh, nullptr, nullptr, nullptr, 0.0f, lane);
    __syncthreads();
    lds_gemm_h<0, true, false, false, true>(s_gh, 64, wg1p, bg1, 64,
                                            nullptr, s_attn, nullptr, nullptr, 0.0f, lane);
    __syncthreads();

    // softmax over 16 neighbors (axis -2), scale 1/sqrt(64), weighted sum
    for (int c = lane; c < 64; c += 32) {
        float mx = -1e30f;
        for (int r = 0; r < 16; ++r) mx = fmaxf(mx, s_attn[r * 64 + c] * 0.125f);
        float e[16];
        float sum = 0.0f;
        for (int r = 0; r < 16; ++r) {
            float ev = __expf(s_attn[r * 64 + c] * 0.125f - mx);
            e[r] = ev; sum += ev;
        }
        float rs = 1.0f / sum;
        float acc = 0.0f;
        for (int r = 0; r < 16; ++r) acc += e[r] * rs * s_vp[r * 64 + c];
        s_res[c] = acc;
    }
    __syncthreads();

    // out = fc2(res) + pre
    for (int o = lane; o < d_points; o += 32) {
        float acc = b2[o];
        for (int c = 0; c < 64; ++c) acc += s_res[c] * w2[(size_t)c * d_points + o];
        out[(size_t)bm * d_points + o] = acc + pre[(size_t)bm * d_points + o];
    }
}

// ---------------------------------------------------------------------------
// Fused transition-down MLP + max-pool: 1 wave per sampled point.
// ---------------------------------------------------------------------------
__global__ void k_transition(const float* __restrict__ refxyz, const float* __restrict__ newxyz,
                             const float* __restrict__ pts, const int* __restrict__ idx,
                             const unsigned int* w0p, const float* b0,
                             const float* g0, const float* be0,
                             const unsigned int* w1p, const float* b1,
                             const float* g1, const float* be1,
                             float* __restrict__ outp, int Nref, int Mnew,
                             int cfeat, int ch, float inv) {
    __shared__ __align__(4) _Float16 s_gh[16 * 160];
    __shared__ __align__(4) _Float16 s_h1h[16 * 256];
    __shared__ float s_h2[16 * 256];
    __shared__ int   s_idx[16];

    int lane = threadIdx.x;
    int bm   = blockIdx.x;
    int b    = bm / Mnew;
    if (lane < 16) s_idx[lane] = idx[(size_t)bm * 16 + lane];
    __syncthreads();

    int cin  = cfeat + 3;
    int kpad = ((cin + 31) / 32) * 32;   // 67->96, 131->160
    float nx = newxyz[(size_t)bm * 3 + 0];
    float ny = newxyz[(size_t)bm * 3 + 1];
    float nz = newxyz[(size_t)bm * 3 + 2];
    for (int t = lane; t < 16 * kpad; t += 32) {
        int r = t / kpad, c = t % kpad;
        float v = 0.0f;
        if (c < cin) {
            int nb = s_idx[r];
            if (c < 3) {
                float ctr = (c == 0) ? nx : (c == 1) ? ny : nz;
                v = refxyz[((size_t)b * Nref + nb) * 3 + c] - ctr;
            } else {
                v = pts[((size_t)b * Nref + nb) * cfeat + (c - 3)];
            }
        }
        s_gh[r * kpad + c] = (_Float16)v;
    }
    __syncthreads();

    lds_gemm_h<1, true, true, true, false>(s_gh, kpad, w0p, b0, ch,
                                           s_h1h, nullptr, g0, be0, inv, lane);
    __syncthreads();
    lds_gemm_h<1, true, true, false, true>(s_h1h, ch, w1p, b1, ch,
                                           nullptr, s_h2, g1, be1, inv, lane);
    __syncthreads();

    for (int c = lane; c < ch; c += 32) {
        float mx = -1e30f;
        for (int r = 0; r < 16; ++r) mx = fmaxf(mx, s_h2[r * ch + c]);
        outp[(size_t)bm * ch + c] = mx;
    }
}

// ---------------------------------------------------------------------------
// Farthest point sampling: one 256-thread block per batch.
// ---------------------------------------------------------------------------
__global__ void k_fps(const float* __restrict__ xyz, float* __restrict__ dists,
                      int* __restrict__ fidx, int N, int npoint) {
    const int T = 256;
    int b = blockIdx.x, tid = threadIdx.x;
    __shared__ float rv[256];
    __shared__ int   ri[256];
    __shared__ int   s_far;
    const float* px = xyz + (size_t)b * N * 3;
    float* dd = dists + (size_t)b * N;
    for (int i = tid; i < N; i += T) dd[i] = 1e10f;
    if (tid == 0) s_far = 0;
    __syncthreads();
    for (int t = 0; t < npoint; ++t) {
        int far = s_far;
        if (tid == 0) fidx[b * npoint + t] = far;
        float cx = px[far * 3], cy = px[far * 3 + 1], cz = px[far * 3 + 2];
        float bv = -1.0f; int bi = 0x7fffffff;
        for (int i = tid; i < N; i += T) {
            float dx = px[i * 3] - cx, dy = px[i * 3 + 1] - cy, dz = px[i * 3 + 2] - cz;
            float d  = dx * dx + dy * dy + dz * dz;
            float nv = fminf(dd[i], d);
            dd[i] = nv;
            if (nv > bv || (nv == bv && i < bi)) { bv = nv; bi = i; }
        }
        rv[tid] = bv; ri[tid] = bi;
        __syncthreads();
        for (int s = 128; s > 0; s >>= 1) {
            if (tid < s) {
                if (rv[tid + s] > rv[tid] ||
                    (rv[tid + s] == rv[tid] && ri[tid + s] < ri[tid])) {
                    rv[tid] = rv[tid + s]; ri[tid] = ri[tid + s];
                }
            }
            __syncthreads();
        }
        if (tid == 0) s_far = ri[0];
        __syncthreads();
    }
}

// ---------------------------------------------------------------------------
// Brute-force KNN (k=16): one thread per query.
// ---------------------------------------------------------------------------
__global__ void k_knn(const float* __restrict__ qxyz, const float* __restrict__ rxyz,
                      int* __restrict__ idx, int M, int Nref, int total) {
    int g = blockIdx.x * blockDim.x + threadIdx.x;
    if (g >= total) return;
    int b = g / M;
    float qx = qxyz[(size_t)g * 3], qy = qxyz[(size_t)g * 3 + 1], qz = qxyz[(size_t)g * 3 + 2];
    float bd[16]; int bi[16];
#pragma unroll
    for (int j = 0; j < 16; ++j) { bd[j] = 1e30f; bi[j] = 0; }
    const float* rp = rxyz + (size_t)b * Nref * 3;
    for (int i = 0; i < Nref; ++i) {
        float dx = rp[i * 3] - qx, dy = rp[i * 3 + 1] - qy, dz = rp[i * 3 + 2] - qz;
        float d = dx * dx + dy * dy + dz * dz;
        if (d < bd[15]) {
            int j = 15;
            while (j > 0 && bd[j - 1] > d) { bd[j] = bd[j - 1]; bi[j] = bi[j - 1]; --j; }
            bd[j] = d; bi[j] = i;
        }
    }
#pragma unroll
    for (int j = 0; j < 16; ++j) idx[(size_t)g * 16 + j] = bi[j];
}

// ---------------------------------------------------------------------------
// Small helpers
// ---------------------------------------------------------------------------
__global__ void k_extract_xyz(const float* __restrict__ x, float* __restrict__ xyz, int total) {
    int i = blockIdx.x * blockDim.x + threadIdx.x;
    if (i < total) {
        xyz[i * 3 + 0] = x[i * 6 + 0];
        xyz[i * 3 + 1] = x[i * 6 + 1];
        xyz[i * 3 + 2] = x[i * 6 + 2];
    }
}

__global__ void k_gather_xyz(const float* __restrict__ xyz, const int* __restrict__ fidx,
                             float* __restrict__ nxyz, int Nref, int Mnew, int total) {
    int g = blockIdx.x * blockDim.x + threadIdx.x;
    if (g < total) {
        int b = g / Mnew;
        int i = fidx[g];
        nxyz[(size_t)g * 3 + 0] = xyz[((size_t)b * Nref + i) * 3 + 0];
        nxyz[(size_t)g * 3 + 1] = xyz[((size_t)b * Nref + i) * 3 + 1];
        nxyz[(size_t)g * 3 + 2] = xyz[((size_t)b * Nref + i) * 3 + 2];
    }
}

__global__ void k_copy(const float* __restrict__ src, float* __restrict__ dst, int n) {
    int i = blockIdx.x * blockDim.x + threadIdx.x;
    if (i < n) dst[i] = src[i];
}

// ---------------------------------------------------------------------------
// Launch.  Input leaf order = setup_inputs() dict insertion order:
//  0:x  1-4:fc1(w,b,w,b)  5-19:t1{fc1 w,b; fc2 w,b; fc_delta w,b,w,b;
//  fc_gamma w,b,w,b; w_qs; w_ks; w_vs}  20-42:block0{sa0 w,b,g,be; sa1 w,b,g,be;
//  tf}  43-65:block1  66-71:head fc2 (w,b)x3
// ---------------------------------------------------------------------------
extern "C" void kernel_launch(void* const* d_in, const int* in_sizes, int n_in,
                              void* d_out, int out_size, void* d_ws, size_t ws_size,
                              hipStream_t stream) {
    (void)in_sizes; (void)n_in; (void)out_size; (void)ws_size;
    auto F = [&](int i) { return (const float*)d_in[i]; };

    const int B = 4, N0 = 4096, M1 = 1024, M2 = 256;
    const float INV = 0.9999950000374997f;  // 1/sqrt(1+1e-5)

    float* wsf = (float*)d_ws;
    size_t off = 0;
    auto alloc = [&](size_t n) { float* p = wsf + off; off += n; return p; };
    float* xyzA = alloc((size_t)B * N0 * 3);
    float* xyzB = alloc((size_t)B * N0 * 3);
    float* ptsA = alloc((size_t)B * N0 * 64);
    float* ptsB = alloc((size_t)B * N0 * 64);
    float* xlin = alloc((size_t)B * N0 * 64);
    float* qb   = alloc((size_t)B * N0 * 64);
    float* kb   = alloc((size_t)B * N0 * 64);
    float* vb   = alloc((size_t)B * N0 * 64);
    float* hid  = alloc((size_t)B * N0 * 64);
    float* fpsd = alloc((size_t)B * N0);
    int*   idxb = (int*)alloc((size_t)B * N0 * 16);
    int*   fidx = (int*)alloc((size_t)B * M1);

    // --- pack all WMMA weights to padded f16 pairs ---
    unsigned int* pw[72] = {};
    auto pack = [&](int idx, int Kd, int N) {
        int Kp = ((Kd + 31) / 32) * 32;
        size_t tot = (size_t)(Kp / 2) * N;
        unsigned int* dst = (unsigned int*)alloc(tot);
        k_pack_w<<<(int)((tot + 255) / 256), 256, 0, stream>>>(F(idx), dst, Kd, N, Kp);
        pw[idx] = dst;
    };
    pack(1, 6, 64);   pack(3, 64, 64);
    pack(5, 64, 64);  pack(17, 64, 64); pack(18, 64, 64); pack(19, 64, 64);
    pack(9, 3, 64);   pack(11, 64, 64); pack(13, 64, 64); pack(15, 64, 64);
    pack(20, 67, 128);  pack(24, 128, 128);
    pack(28, 128, 64);  pack(40, 64, 64); pack(41, 64, 64); pack(42, 64, 64);
    pack(32, 3, 64);    pack(34, 64, 64); pack(36, 64, 64); pack(38, 64, 64);
    pack(43, 131, 256); pack(47, 256, 256);
    pack(51, 256, 64);  pack(63, 64, 64); pack(64, 64, 64); pack(65, 64, 64);
    pack(55, 3, 64);    pack(57, 64, 64); pack(59, 64, 64); pack(61, 64, 64);
    pack(66, 256, 256); pack(68, 256, 256); pack(70, 256, 64);

    auto gemm = [&](const float* X, int widx, const float* bias, float* Y,
                    int R, int Kd, int N, int act) {
        int Kp = ((Kd + 31) / 32) * 32;
        dim3 g((R + 15) / 16, N / 64);
        dim3 t(32);
        if (act) {
            if (bias) k_gemm<1, 1><<<g, t, 0, stream>>>(X, pw[widx], bias, Y, R, Kd, Kp, N);
            else      k_gemm<1, 0><<<g, t, 0, stream>>>(X, pw[widx], bias, Y, R, Kd, Kp, N);
        } else {
            if (bias) k_gemm<0, 1><<<g, t, 0, stream>>>(X, pw[widx], bias, Y, R, Kd, Kp, N);
            else      k_gemm<0, 0><<<g, t, 0, stream>>>(X, pw[widx], bias, Y, R, Kd, Kp, N);
        }
    };

    int totN = B * N0;
    k_extract_xyz<<<(totN + 255) / 256, 256, 0, stream>>>(F(0), xyzA, totN);

    // fc1 MLP: (B*N,6) -> relu -> 64 -> 64
    gemm(F(0), 1, F(2), hid, totN, 6, 64, 1);
    gemm(hid, 3, F(4), ptsA, totN, 64, 64, 0);

    // t1 transformer
    gemm(ptsA, 5, F(6), xlin, totN, 64, 64, 0);
    gemm(xlin, 17, nullptr, qb, totN, 64, 64, 0);
    gemm(xlin, 18, nullptr, kb, totN, 64, 64, 0);
    gemm(xlin, 19, nullptr, vb, totN, 64, 64, 0);
    k_knn<<<(totN + 127) / 128, 128, 0, stream>>>(xyzA, xyzA, idxb, N0, N0, totN);
    k_transformer<<<totN, 32, 0, stream>>>(xyzA, idxb, qb, kb, vb, ptsA,
        pw[9], F(10), pw[11], F(12), pw[13], F(14), pw[15], F(16), F(7), F(8),
        ptsB, N0, 64);

    // block 0: transition down 4096 -> 1024, ch=128
    k_fps<<<B, 256, 0, stream>>>(xyzA, fpsd, fidx, N0, M1);
    k_gather_xyz<<<(B * M1 + 255) / 256, 256, 0, stream>>>(xyzA, fidx, xyzB, N0, M1, B * M1);
    k_knn<<<(B * M1 + 127) / 128, 128, 0, stream>>>(xyzB, xyzA, idxb, M1, N0, B * M1);
    k_transition<<<B * M1, 32, 0, stream>>>(xyzA, xyzB, ptsB, idxb,
        pw[20], F(21), F(22), F(23), pw[24], F(25), F(26), F(27),
        ptsA, N0, M1, 64, 128, INV);
    // block 0 transformer (d_points=128)
    gemm(ptsA, 28, F(29), xlin, B * M1, 128, 64, 0);
    gemm(xlin, 40, nullptr, qb, B * M1, 64, 64, 0);
    gemm(xlin, 41, nullptr, kb, B * M1, 64, 64, 0);
    gemm(xlin, 42, nullptr, vb, B * M1, 64, 64, 0);
    k_knn<<<(B * M1 + 127) / 128, 128, 0, stream>>>(xyzB, xyzB, idxb, M1, M1, B * M1);
    k_transformer<<<B * M1, 32, 0, stream>>>(xyzB, idxb, qb, kb, vb, ptsA,
        pw[32], F(33), pw[34], F(35), pw[36], F(37), pw[38], F(39), F(30), F(31),
        ptsB, M1, 128);

    // block 1: transition down 1024 -> 256, ch=256
    k_fps<<<B, 256, 0, stream>>>(xyzB, fpsd, fidx, M1, M2);
    k_gather_xyz<<<(B * M2 + 255) / 256, 256, 0, stream>>>(xyzB, fidx, xyzA, M1, M2, B * M2);
    k_knn<<<(B * M2 + 127) / 128, 128, 0, stream>>>(xyzA, xyzB, idxb, M2, M1, B * M2);
    k_transition<<<B * M2, 32, 0, stream>>>(xyzB, xyzA, ptsB, idxb,
        pw[43], F(44), F(45), F(46), pw[47], F(48), F(49), F(50),
        ptsA, M1, M2, 128, 256, INV);
    // block 1 transformer (d_points=256)
    gemm(ptsA, 51, F(52), xlin, B * M2, 256, 64, 0);
    gemm(xlin, 63, nullptr, qb, B * M2, 64, 64, 0);
    gemm(xlin, 64, nullptr, kb, B * M2, 64, 64, 0);
    gemm(xlin, 65, nullptr, vb, B * M2, 64, 64, 0);
    k_knn<<<(B * M2 + 127) / 128, 128, 0, stream>>>(xyzA, xyzA, idxb, M2, M2, B * M2);
    k_transformer<<<B * M2, 32, 0, stream>>>(xyzA, idxb, qb, kb, vb, ptsA,
        pw[55], F(56), pw[57], F(58), pw[59], F(60), pw[61], F(62), F(53), F(54),
        ptsB, M2, 256);

    // head: 256 -> relu 256 -> relu 256 -> 64
    gemm(ptsB, 66, F(67), ptsA, B * M2, 256, 256, 1);
    gemm(ptsA, 68, F(69), xlin, B * M2, 256, 256, 1);
    gemm(xlin, 70, F(71), (float*)d_out, B * M2, 256, 64, 0);

    // append final xyz
    k_copy<<<(B * M2 * 3 + 255) / 256, 256, 0, stream>>>(
        xyzA, (float*)d_out + (size_t)B * M2 * 64, B * M2 * 3);
}
